// Predictor_35330400977230
// MI455X (gfx1250) — matrix operand order repaired and checked by reading, not compiled, not run on hardware
//
#include <hip/hip_runtime.h>
#include <hip/hip_bf16.h>

typedef __bf16 bf16_t;
typedef __attribute__((ext_vector_type(16))) __bf16 v16bf;
typedef __attribute__((ext_vector_type(8)))  float  v8f;

#define GAT_N   512
#define GAT_B   64
#define NEGV    (-9.0e15f)
#define ALPHA_L 0.2f

// ---------------------------------------------------------------------------
// Convert f32 -> bf16 (for Wc, W_out)
// ---------------------------------------------------------------------------
__global__ void cvt_bf16(const float* __restrict__ in, bf16_t* __restrict__ out, int n) {
    int i = blockIdx.x * blockDim.x + threadIdx.x;
    if (i < n) out[i] = (bf16_t)in[i];
}

// ---------------------------------------------------------------------------
// Per-row projection for the skinny layers (Fin=34): Wh = h @ W (Fout=128),
// plus src/dst logit vectors. One wave per row; lane owns features
// {lane, lane+32, lane+64, lane+96}. Stores Wh bf16 row-major [row,128].
// ---------------------------------------------------------------------------
__global__ void gat_prep(const float* __restrict__ h, int Fin,
                         const float* __restrict__ W,   // [Fin,128]
                         const float* __restrict__ a,   // [256]: a1 | a2
                         bf16_t* __restrict__ Whb,      // [BN,128] bf16
                         float* __restrict__ src, float* __restrict__ dst) {
    int lane = threadIdx.x & 31;
    int row  = blockIdx.x * (blockDim.x >> 5) + (threadIdx.x >> 5);
    const float* hp = h + (size_t)row * Fin;
    int f0 = lane, f1 = lane + 32, f2 = lane + 64, f3 = lane + 96;
    float a0 = 0.f, a1v = 0.f, a2v = 0.f, a3 = 0.f;
    for (int k = 0; k < Fin; ++k) {
        float hv = hp[k];
        const float* wr = W + (size_t)k * 128;
        a0  = fmaf(hv, wr[f0], a0);
        a1v = fmaf(hv, wr[f1], a1v);
        a2v = fmaf(hv, wr[f2], a2v);
        a3  = fmaf(hv, wr[f3], a3);
    }
    size_t base = (size_t)row * 128;
    Whb[base + f0] = (bf16_t)a0;
    Whb[base + f1] = (bf16_t)a1v;
    Whb[base + f2] = (bf16_t)a2v;
    Whb[base + f3] = (bf16_t)a3;
    float s = a0 * a[f0] + a1v * a[f1] + a2v * a[f2] + a3 * a[f3];
    float d = a0 * a[128 + f0] + a1v * a[128 + f1] + a2v * a[128 + f2] + a3 * a[128 + f3];
    #pragma unroll
    for (int off = 16; off > 0; off >>= 1) {
        s += __shfl_xor(s, off, 32);
        d += __shfl_xor(d, off, 32);
    }
    if (lane == 0) { src[row] = s; dst[row] = d; }
}

// ---------------------------------------------------------------------------
// WMMA projection for the wide output layer: Wh = multi(bf16,[BN,384]) @
// W_out(bf16,[384,128]), plus src/dst. One wave per 16-row tile; 8 C tiles,
// K-loop 12 x 32, double-buffered B operands.
// ---------------------------------------------------------------------------
__global__ void __launch_bounds__(128) gat_prep_wmma(
        const bf16_t* __restrict__ hb,   // [BN,384] bf16
        const bf16_t* __restrict__ Wb,   // [384,128] bf16
        const float*  __restrict__ a,    // [256] f32
        bf16_t* __restrict__ Whb,        // [BN,128] bf16
        float* __restrict__ src, float* __restrict__ dst) {
    int lane  = threadIdx.x & 31;
    int mtile = blockIdx.x * (blockDim.x >> 5) + (threadIdx.x >> 5);
    int rowA  = lane & 15;
    int half  = lane >> 4;

    v8f acc[8];
    #pragma unroll
    for (int ft = 0; ft < 8; ++ft) acc[ft] = (v8f)0.f;

    const bf16_t* arow = hb + (size_t)(mtile * 16 + rowA) * 384;
    for (int kt = 0; kt < 12; ++kt) {
        const bf16_t* pb = Wb + (size_t)(kt * 32 + half * 8) * 128 + rowA;
        v16bf bcur;                                  // issue first B load early
        #pragma unroll
        for (int i = 0; i < 16; ++i) bcur[i] = pb[((i & 7) + ((i >> 3) << 4)) * 128];
        v16bf av;
        const bf16_t* pa = arow + kt * 32 + half * 8;
        #pragma unroll
        for (int i = 0; i < 8; ++i) av[i] = pa[i];
        #pragma unroll
        for (int i = 0; i < 8; ++i) av[8 + i] = pa[16 + i];
        #pragma unroll
        for (int ft = 0; ft < 8; ++ft) {
            v16bf bnxt = bcur;
            if (ft < 7) {
                #pragma unroll
                for (int i = 0; i < 16; ++i) bnxt[i] = pb[(ft + 1) * 16 + ((i & 7) + ((i >> 3) << 4)) * 128];
            }
            acc[ft] = __builtin_amdgcn_wmma_f32_16x16x32_bf16(
                false, av, false, bcur, (short)0, acc[ft], false, false);
            bcur = bnxt;
        }
    }
    // store Wh (bf16) and accumulate src/dst partials per row
    float sp[8], dp[8];
    #pragma unroll
    for (int r = 0; r < 8; ++r) { sp[r] = 0.f; dp[r] = 0.f; }
    #pragma unroll
    for (int ft = 0; ft < 8; ++ft) {
        float w1 = a[ft * 16 + rowA];
        float w2 = a[128 + ft * 16 + rowA];
        #pragma unroll
        for (int r = 0; r < 8; ++r) {
            float v = acc[ft][r];
            Whb[(size_t)(mtile * 16 + r + half * 8) * 128 + ft * 16 + rowA] = (bf16_t)v;
            sp[r] = fmaf(v, w1, sp[r]);
            dp[r] = fmaf(v, w2, dp[r]);
        }
    }
    #pragma unroll
    for (int off = 1; off < 16; off <<= 1) {
        #pragma unroll
        for (int r = 0; r < 8; ++r) {
            sp[r] += __shfl_xor(sp[r], off, 32);
            dp[r] += __shfl_xor(dp[r], off, 32);
        }
    }
    if (rowA == 0) {
        #pragma unroll
        for (int r = 0; r < 8; ++r) {
            int row = mtile * 16 + r + half * 8;
            src[row] = sp[r];
            dst[row] = dp[r];
        }
    }
}

// ---------------------------------------------------------------------------
// GAT attention layer, single pass. Row-max replaced by the provable upper
// bound M_i = leakyrelu(src_i + max_j dst_j). Probabilities are computed
// directly in the WMMA A-operand lane layout; B operands double-buffered so
// exp VALU + matrix-core execution hide the Wh load latency.
// ---------------------------------------------------------------------------
__global__ void __launch_bounds__(128) gat_attn(
        const bf16_t* __restrict__ Whb,    // [B*N,128] bf16
        const float* __restrict__ src,     // [B*N]
        const float* __restrict__ dst,     // [B*N]
        const int*   __restrict__ adj,     // [B,N,N]
        bf16_t* __restrict__ out, int outStride, int colBase) {
    const int N = GAT_N;
    int lane = threadIdx.x & 31;
    int tile = blockIdx.x * (blockDim.x >> 5) + (threadIdx.x >> 5);
    int b    = tile >> 5;                              // 32 row tiles per batch
    int i0   = (tile & 31) << 4;
    int rowA = lane & 15;
    int half = lane >> 4;

    const float*  dstB   = dst + (size_t)b * N;
    const bf16_t* WhB    = Whb + (size_t)b * N * 128;
    const int*    adjRow = adj + ((size_t)b * N + (i0 + rowA)) * N;

    // per-batch max of dst (upper bound for every row's logit max)
    float md = -3.0e38f;
    #pragma unroll
    for (int t = 0; t < 16; ++t) md = fmaxf(md, dstB[lane + 32 * t]);
    #pragma unroll
    for (int off = 16; off > 0; off >>= 1) md = fmaxf(md, __shfl_xor(md, off, 32));

    float s_i = src[(size_t)b * N + i0 + rowA];
    float m = s_i + md;
    m = (m > 0.f) ? m : (ALPHA_L * m);                 // fixed per-row max bound

    v8f acc[8];
    #pragma unroll
    for (int ft = 0; ft < 8; ++ft) acc[ft] = (v8f)0.f;
    float s = 0.f;

    for (int jt = 0; jt < N / 32; ++jt) {
        int j0 = jt * 32;
        const bf16_t* Bbase = WhB + (size_t)(j0 + half * 8) * 128 + rowA;
        v16bf bcur;                                    // issue first B load early
        #pragma unroll
        for (int i = 0; i < 16; ++i) bcur[i] = Bbase[((i & 7) + ((i >> 3) << 4)) * 128];
        if (jt + 1 < N / 32)
            __builtin_prefetch(adjRow + j0 + 32 + 8 * half, 0, 1);
        // probabilities directly in A-operand lane layout (overlaps B load)
        v16bf av;
        float psum = 0.f;
        #pragma unroll
        for (int i = 0; i < 16; ++i) {
            int jl = (i & 7) + ((i >> 3) << 4) + (half << 3);
            int j  = j0 + jl;
            float e = s_i + dstB[j];
            e = (e > 0.f) ? e : (ALPHA_L * e);
            float v = (adjRow[j] > 0) ? e : NEGV;
            float p = __expf(v - m);
            psum += p;
            av[i] = (bf16_t)p;
        }
        psum += __shfl_xor(psum, 16, 32);
        s += psum;
        // WMMA: probs(16x32) @ Wh(32x128); B-operand column = lane%16, K in table order
        #pragma unroll
        for (int ft = 0; ft < 8; ++ft) {
            v16bf bnxt = bcur;
            if (ft < 7) {
                #pragma unroll
                for (int i = 0; i < 16; ++i) bnxt[i] = Bbase[(ft + 1) * 16 + ((i & 7) + ((i >> 3) << 4)) * 128];
            }
            acc[ft] = __builtin_amdgcn_wmma_f32_16x16x32_bf16(
                false, av, false, bcur, (short)0, acc[ft], false, false);
            bcur = bnxt;
        }
    }

    // normalize, elu, store bf16
    float inv = 1.0f / s;
    #pragma unroll
    for (int r = 0; r < 8; ++r) {
        float flo = __shfl(inv, r, 32);
        float fhi = __shfl(inv, r + 8, 32);
        float f = half ? fhi : flo;
        int grow = i0 + r + half * 8;
        #pragma unroll
        for (int ft = 0; ft < 8; ++ft) {
            float hv = acc[ft][r] * f;
            hv = (hv > 0.f) ? hv : (__expf(hv) - 1.0f);   // elu
            out[((size_t)b * N + grow) * outStride + colBase + ft * 16 + rowA] = (bf16_t)hv;
        }
    }
}

// ---------------------------------------------------------------------------
// Final linear: out = leaky_relu(x @ Wc + bc), x:[BN,128] bf16, Wc:[128,256].
// One wave: one 16-row tile x 64 cols (4 WMMA C tiles), K-loop of 4x32,
// double-buffered B operands.
// ---------------------------------------------------------------------------
__global__ void __launch_bounds__(128) final_linear(
        const bf16_t* __restrict__ xb,
        const bf16_t* __restrict__ Wcb,
        const float*  __restrict__ bc,
        float* __restrict__ out) {
    int lane = threadIdx.x & 31;
    int wid  = blockIdx.x * (blockDim.x >> 5) + (threadIdx.x >> 5);
    int mtile = wid >> 2;
    int ng    = wid & 3;
    int rowA = lane & 15;
    int half = lane >> 4;

    v8f acc[4];
    #pragma unroll
    for (int nt = 0; nt < 4; ++nt) acc[nt] = (v8f)0.f;

    const bf16_t* xrow = xb + (size_t)(mtile * 16 + rowA) * 128;
    for (int kt = 0; kt < 4; ++kt) {
        const bf16_t* pb = Wcb + (size_t)(kt * 32 + half * 8) * 256 + ng * 64 + rowA;
        v16bf bcur;                                    // issue first B load early
        #pragma unroll
        for (int i = 0; i < 16; ++i) bcur[i] = pb[((i & 7) + ((i >> 3) << 4)) * 256];
        v16bf av;
        const bf16_t* pa = xrow + kt * 32 + half * 8;
        #pragma unroll
        for (int i = 0; i < 8; ++i) av[i] = pa[i];          // K 0..7 (+8*half)
        #pragma unroll
        for (int i = 0; i < 8; ++i) av[8 + i] = pa[16 + i]; // K 16..23 (+8*half)
        #pragma unroll
        for (int nt = 0; nt < 4; ++nt) {
            v16bf bnxt = bcur;
            if (nt < 3) {
                #pragma unroll
                for (int i = 0; i < 16; ++i) bnxt[i] = pb[(nt + 1) * 16 + ((i & 7) + ((i >> 3) << 4)) * 256];
            }
            acc[nt] = __builtin_amdgcn_wmma_f32_16x16x32_bf16(
                false, av, false, bcur, (short)0, acc[nt], false, false);
            bcur = bnxt;
        }
    }
    #pragma unroll
    for (int nt = 0; nt < 4; ++nt) {
        int col = ng * 64 + nt * 16 + rowA;
        float bias = bc[col];
        #pragma unroll
        for (int r = 0; r < 8; ++r) {
            int grow = mtile * 16 + r + half * 8;
            float v = acc[nt][r] + bias;
            v = (v > 0.f) ? v : (ALPHA_L * v);
            out[(size_t)grow * 256 + col] = v;
        }
    }
}

// ---------------------------------------------------------------------------
extern "C" void kernel_launch(void* const* d_in, const int* in_sizes, int n_in,
                              void* d_out, int out_size, void* d_ws, size_t ws_size,
                              hipStream_t stream) {
    const float* compound = (const float*)d_in[0];   // [64,512,34]
    const int*   adj      = (const int*)  d_in[1];   // [64,512,512]
    const float* W_stack  = (const float*)d_in[2];   // [3,34,128]
    const float* a_stack  = (const float*)d_in[3];   // [3,256,1]
    const float* W_out    = (const float*)d_in[4];   // [384,128]
    const float* a_out    = (const float*)d_in[5];   // [256,1]
    const float* Wc       = (const float*)d_in[6];   // [128,256]
    const float* bc       = (const float*)d_in[7];   // [256]
    float* out = (float*)d_out;

    const int BN = GAT_B * GAT_N;                    // 32768
    char* ws = (char*)d_ws;
    bf16_t* Whb    = (bf16_t*)(ws);                  //  8 MB  [BN,128]
    float*  src    = (float*) (ws + 8388608);        // 128 KB
    float*  dstv   = (float*) (ws + 8519680);        // 128 KB
    bf16_t* multiB = (bf16_t*)(ws + 8650752);        // 24 MB  [BN,384] bf16
    bf16_t* xb     = (bf16_t*)(ws + 33816576);       //  8 MB  [BN,128] bf16
    bf16_t* Wcb    = (bf16_t*)(ws + 42205184);       // 64 KB
    bf16_t* Woutb  = (bf16_t*)(ws + 42270720);       // 96 KB  [384,128] bf16

    cvt_bf16<<<(128 * 256 + 255) / 256, 256, 0, stream>>>(Wc, Wcb, 128 * 256);
    cvt_bf16<<<(384 * 128 + 255) / 256, 256, 0, stream>>>(W_out, Woutb, 384 * 128);

    const int prepBlocks = BN / 8;        // 8 waves/block, 1 row/wave
    const int attnBlocks = (BN / 16) / 4; // 4 waves/block, 1 tile/wave

    for (int l = 0; l < 3; ++l) {
        gat_prep<<<prepBlocks, 256, 0, stream>>>(
            compound, 34, W_stack + (size_t)l * 34 * 128, a_stack + (size_t)l * 256,
            Whb, src, dstv);
        gat_attn<<<attnBlocks, 128, 0, stream>>>(
            Whb, src, dstv, adj, multiB, 384, l * 128);
    }
    // output GAT layer on concatenated heads (WMMA projection)
    gat_prep_wmma<<<(BN / 16) / 4, 128, 0, stream>>>(
        multiB, Woutb, a_out, Whb, src, dstv);
    gat_attn<<<attnBlocks, 128, 0, stream>>>(
        Whb, src, dstv, adj, xb, 128, 0);
    // final linear + leaky relu
    final_linear<<<(BN / 16), 128, 0, stream>>>(xb, Wcb, bc, out);
}
